// ReconfiguredAttentionModule_27058293965150
// MI455X (gfx1250) — matrix operand order repaired and checked by reading, not compiled, not run on hardware
//
#include <hip/hip_runtime.h>
#include <hip/hip_bf16.h>

typedef _Float16 h16;
typedef __attribute__((ext_vector_type(16))) _Float16 v16h;
typedef __attribute__((ext_vector_type(8)))  _Float16 v8h;
typedef __attribute__((ext_vector_type(8)))  float    v8f;

#define C_      256
#define TC      768
#define NPATCH  12544     // 112*112
#define TPB     12644     // tokens per batch (12544 + 100)
#define NROWS   101152    // 8 * 12644
#define MDET    100
#define OUT1    25690112  // 8*12544*256

__device__ __forceinline__ v8h ld8(const h16* p) { return *(const v8h*)p; }
__device__ __forceinline__ v16h cat8(v8h lo, v8h hi) {
  return __builtin_shufflevector(lo, hi, 0,1,2,3,4,5,6,7,8,9,10,11,12,13,14,15);
}
__device__ __forceinline__ v8f wmma_f16(v16h a, v16h b, v8f c) {
  return __builtin_amdgcn_wmma_f32_16x16x32_f16(false, a, false, b, (short)0, c, false, false);
}
__device__ __forceinline__ v8f vzero8() { v8f z = {0.f,0.f,0.f,0.f,0.f,0.f,0.f,0.f}; return z; }

// LDS byte offset of a __shared__ object: flat LDS addresses carry the LDS
// offset in addr[31:0] (ISA 10.2 aperture mapping), so truncate.
__device__ __forceinline__ unsigned lds_off(const void* p) {
  return (unsigned)(unsigned long long)p;
}
// CDNA5 async global->LDS copy (ASYNCcnt-tracked, bypasses VGPRs).
__device__ __forceinline__ void async_ld_b128(unsigned lds, const void* g) {
  asm volatile("global_load_async_to_lds_b128 %0, %1, off"
               :: "v"(lds), "v"(g) : "memory");
}
__device__ __forceinline__ void wait_async_le3() {
  asm volatile("s_wait_asynccnt 0x3" ::: "memory");
}
__device__ __forceinline__ void wait_async_le1() {
  asm volatile("s_wait_asynccnt 0x1" ::: "memory");
}
__device__ __forceinline__ void wait_async_0() {
  asm volatile("s_wait_asynccnt 0x0" ::: "memory");
}

// ---------------------------------------------------------------------------
// Kernel 0: weights -> f16, transposed:  Wt[c][k] = (h16)W[k][c]
// ---------------------------------------------------------------------------
__global__ __launch_bounds__(256) void prep_weights(
    const float* __restrict__ qkv_w, const float* __restrict__ proj_w,
    h16* __restrict__ wq_t, h16* __restrict__ wp_t)
{
  int idx = blockIdx.x * 256 + threadIdx.x;
  if (idx < TC * C_) {                       // 768*256
    int c = idx >> 8, k = idx & 255;
    wq_t[idx] = (h16)qkv_w[(size_t)k * TC + c];
  } else {
    int j = idx - TC * C_;
    if (j < C_ * C_) {
      int c = j >> 8, k = j & 255;
      wp_t[j] = (h16)proj_w[(size_t)k * C_ + c];
    }
  }
}

// ---------------------------------------------------------------------------
// Kernel 1: qkv = concat(x, det) @ qkv_w + qkv_b   (store f16)
// 128x64 tile / 256 threads. B tile: async-to-LDS, double buffered (1 async
// instr per wave per tile -> s_wait_asynccnt 0x1 overlaps next-tile fetch).
// A tile: fp32 gather + convert (input is fp32), with prefetch of next slice.
// ---------------------------------------------------------------------------
__global__ __launch_bounds__(256) void qkv_gemm(
    const float* __restrict__ x, const float* __restrict__ det,
    const h16* __restrict__ wq_t, const float* __restrict__ bias,
    h16* __restrict__ qkv)
{
  __shared__ __align__(16) h16 As[128][40];      // rows x k   (pitch 80B)
  __shared__ __align__(16) h16 BsT[2][64][40];   // cols x k, double buffered
  const int tid  = threadIdx.x;
  const int lane = tid & 31, wave = tid >> 5;
  const int r0 = blockIdx.y * 128;
  const int c0 = blockIdx.x * 64;
  const int m  = lane & 15, h = lane >> 4;

  // B issue: 64 rows x 32 halves = 4KB = 256 threads x one b128
  const int bc = tid >> 2, bseg = tid & 3;
  const unsigned bdst[2] = {
    lds_off(&BsT[0][bc][bseg * 8]), lds_off(&BsT[1][bc][bseg * 8]) };
  const h16* bsrc = wq_t + (size_t)(c0 + bc) * C_ + bseg * 8;

  v8f acc[4];
  for (int i = 0; i < 4; ++i) acc[i] = vzero8();

  async_ld_b128(bdst[0], bsrc);                 // tile 0 B
  for (int step = 0; step < 8; ++step) {
    const int kk0 = step * 32;
    __syncthreads();                            // prior tile reads finished
    if (step < 7)
      async_ld_b128(bdst[(step + 1) & 1], bsrc + kk0 + 32);
    // --- A tile: 128 rows x 32 k (fp32 -> f16) ---
    for (int p = 0; p < 4; ++p) {
      int idx = tid + p * 256;                  // 0..1023
      int row = idx >> 3, seg = idx & 7;        // 8 x float4 per row
      int rg  = r0 + row;
      float4 v = make_float4(0.f, 0.f, 0.f, 0.f);
      if (rg < NROWS) {
        int b = rg / TPB, t = rg - b * TPB;
        const float* src = (t < NPATCH)
            ? (x   + ((size_t)b * NPATCH + t) * C_)
            : (det + ((size_t)b * MDET + (t - NPATCH)) * C_);
        v = *(const float4*)(src + kk0 + seg * 4);
        if (step < 7) __builtin_prefetch(src + kk0 + 32 + seg * 4, 0, 0);
      }
      h16* dp = &As[row][seg * 4];
      dp[0] = (h16)v.x; dp[1] = (h16)v.y; dp[2] = (h16)v.z; dp[3] = (h16)v.w;
    }
    if (step < 7) wait_async_le1(); else wait_async_0();  // tile `step` B landed
    __syncthreads();
    // --- fragments + WMMA ---
    v16h a = cat8(ld8(&As[wave * 16 + m][16 * h]),
                  ld8(&As[wave * 16 + m][16 * h + 8]));
    const int buf = step & 1;
    for (int ci = 0; ci < 4; ++ci) {
      v16h bf = cat8(ld8(&BsT[buf][ci * 16 + m][16 * h]),
                     ld8(&BsT[buf][ci * 16 + m][16 * h + 8]));
      acc[ci] = wmma_f16(a, bf, acc[ci]);
    }
  }
  // --- epilogue: + bias, store f16 ---
  for (int r = 0; r < 8; ++r) {
    int row = r0 + wave * 16 + r + 8 * h;
    if (row >= NROWS) continue;
    for (int ci = 0; ci < 4; ++ci) {
      int col = c0 + ci * 16 + m;
      qkv[(size_t)row * TC + col] = (h16)(acc[ci][r] + bias[col]);
    }
  }
}

// ---------------------------------------------------------------------------
// Kernel 2: windowed attention (64 tokens, hd=32) per (window, head).
// Block = 64 threads = 2 waves; each wave = one head. Grid = 1568 * 4.
// ---------------------------------------------------------------------------
__global__ __launch_bounds__(64) void win_attn(
    const h16* __restrict__ qkv, const float* __restrict__ mask,
    const float* __restrict__ rpb, h16* __restrict__ att)
{
  __shared__ __align__(16) h16 PB[2][64 * 72];  // bias+mask, later P (per wave)
  __shared__ __align__(16) h16 VT[2][32 * 72];  // V transposed [d][j] (per wave)

  const int tid  = threadIdx.x;
  const int lane = tid & 31, wave = tid >> 5;
  const int wdw  = blockIdx.x >> 2;             // global window 0..1567
  const int head = (blockIdx.x & 3) * 2 + wave;
  const int b    = wdw / 196, w = wdw - b * 196;
  const int wh   = w / 14,  ww = w - wh * 14;
  const size_t rowbase = (size_t)b * TPB;
  const int m = lane & 15, h = lane >> 4;
  const float scale = 0.17677669529663687f;     // 1/sqrt(32)

  h16* PBh = PB[wave];
  h16* VTh = VT[wave];

  auto trow = [&](int i) { return (wh * 8 + (i >> 3)) * 112 + ww * 8 + (i & 7); };

  // 1) relative-position bias + window mask -> PBh (f16)
  {
    const size_t mbase = (size_t)w * 4096;
    for (int idx = lane; idx < 4096; idx += 32) {
      int i = idx >> 6, j = idx & 63;
      int ridx = ((i >> 3) - (j >> 3) + 7) * 15 + ((i & 7) - (j & 7) + 7);
      PBh[i * 72 + j] = (h16)(rpb[ridx * 8 + head] + mask[mbase + idx]);
    }
  }
  // 2) V (64x32) transposed into LDS: VTh[d][j]
  for (int jj = lane; jj < 64; jj += 32) {
    const h16* vp = qkv + (rowbase + trow(jj)) * TC + 2 * C_ + head * 32;
    v8h a0 = ld8(vp), a1 = ld8(vp + 8), a2 = ld8(vp + 16), a3 = ld8(vp + 24);
    for (int d = 0; d < 8; ++d) {
      VTh[(d     ) * 72 + jj] = a0[d];
      VTh[(d +  8) * 72 + jj] = a1[d];
      VTh[(d + 16) * 72 + jj] = a2[d];
      VTh[(d + 24) * 72 + jj] = a3[d];
    }
  }
  __syncthreads();

  // K B-fragments (4 col tiles), direct from global (contiguous 32B per lane)
  v16h kb[4];
  for (int ci = 0; ci < 4; ++ci) {
    const h16* kp = qkv + (rowbase + trow(ci * 16 + m)) * TC + C_ + head * 32 + 16 * h;
    kb[ci] = cat8(ld8(kp), ld8(kp + 8));
  }

  float rinv[4][8];
  // 3) S = scale*(Q K^T) + bias+mask, softmax per row, P -> PBh (overwrites bias)
  for (int ri = 0; ri < 4; ++ri) {
    const h16* qp = qkv + (rowbase + trow(ri * 16 + m)) * TC + head * 32;
    v16h qa = cat8(ld8(qp + 16 * h), ld8(qp + 16 * h + 8));
    v8f s[4];
    for (int ci = 0; ci < 4; ++ci) s[ci] = wmma_f16(qa, kb[ci], vzero8());
    for (int ci = 0; ci < 4; ++ci)
      for (int r = 0; r < 8; ++r) {
        int i = ri * 16 + r + 8 * h, j = ci * 16 + m;
        s[ci][r] = s[ci][r] * scale + (float)PBh[i * 72 + j];
      }
    for (int r = 0; r < 8; ++r) {
      float mx = -1e30f;
      for (int ci = 0; ci < 4; ++ci) mx = fmaxf(mx, s[ci][r]);
      for (int o = 8; o; o >>= 1) mx = fmaxf(mx, __shfl_xor(mx, o, 32));
      float sum = 0.f;
      for (int ci = 0; ci < 4; ++ci) {
        float e = __expf(s[ci][r] - mx);
        s[ci][r] = e; sum += e;
      }
      for (int o = 8; o; o >>= 1) sum += __shfl_xor(sum, o, 32);
      rinv[ri][r] = 1.0f / sum;
    }
    for (int ci = 0; ci < 4; ++ci)
      for (int r = 0; r < 8; ++r) {
        int i = ri * 16 + r + 8 * h, j = ci * 16 + m;
        PBh[i * 72 + j] = (h16)s[ci][r];       // unnormalized P
      }
  }
  __syncthreads();   // cross-lane visibility of P / VT

  // V B-fragments from VTh: (kc = j-chunk, ti = d-tile)
  v16h vb[2][2];
  for (int kc = 0; kc < 2; ++kc)
    for (int ti = 0; ti < 2; ++ti) {
      const h16* p = VTh + (ti * 16 + m) * 72 + kc * 32 + 16 * h;
      vb[kc][ti] = cat8(ld8(p), ld8(p + 8));
    }

  // 4) O = P V, normalize by 1/rowsum, scatter to att (f16)
  for (int ri = 0; ri < 4; ++ri) {
    v8f o0 = vzero8(), o1 = vzero8();
    for (int kc = 0; kc < 2; ++kc) {
      const h16* pp = PBh + (ri * 16 + m) * 72 + kc * 32;
      v16h pa = cat8(ld8(pp + 16 * h), ld8(pp + 16 * h + 8));
      o0 = wmma_f16(pa, vb[kc][0], o0);
      o1 = wmma_f16(pa, vb[kc][1], o1);
    }
    for (int r = 0; r < 8; ++r) {
      int i = ri * 16 + r + 8 * h;
      size_t row = rowbase + trow(i);
      float rv = rinv[ri][r];
      att[row * C_ + head * 32 +      m] = (h16)(o0[r] * rv);
      att[row * C_ + head * 32 + 16 + m] = (h16)(o1[r] * rv);
    }
  }
}

// ---------------------------------------------------------------------------
// Kernel 3: det-token attention (100 tokens, hd=32), scalar (41 MFLOP total)
// ---------------------------------------------------------------------------
__global__ __launch_bounds__(128) void det_attn(
    const h16* __restrict__ qkv, h16* __restrict__ att)
{
  __shared__ h16 Ks[MDET * 32];
  __shared__ h16 Vs[MDET * 32];
  const int b = blockIdx.x >> 3, head = blockIdx.x & 7;
  const size_t base = (size_t)b * TPB + NPATCH;
  const float scale = 0.17677669529663687f;

  for (int idx = threadIdx.x; idx < MDET * 32; idx += 128) {
    int j = idx >> 5, d = idx & 31;
    size_t row = base + j;
    Ks[idx] = qkv[row * TC +     C_ + head * 32 + d];
    Vs[idx] = qkv[row * TC + 2 * C_ + head * 32 + d];
  }
  __syncthreads();

  int i = threadIdx.x;
  if (i < MDET) {
    float q[32];
    const h16* qp = qkv + (base + i) * TC + head * 32;
    for (int d = 0; d < 32; ++d) q[d] = (float)qp[d] * scale;
    float mx = -1e30f;
    for (int j = 0; j < MDET; ++j) {
      float s = 0.f;
      for (int d = 0; d < 32; ++d) s += q[d] * (float)Ks[j * 32 + d];
      mx = fmaxf(mx, s);
    }
    float sum = 0.f, o[32];
    for (int d = 0; d < 32; ++d) o[d] = 0.f;
    for (int j = 0; j < MDET; ++j) {
      float s = 0.f;
      for (int d = 0; d < 32; ++d) s += q[d] * (float)Ks[j * 32 + d];
      float p = __expf(s - mx);
      sum += p;
      for (int d = 0; d < 32; ++d) o[d] += p * (float)Vs[j * 32 + d];
    }
    float rv = 1.f / sum;
    h16* op = att + (base + i) * C_ + head * 32;
    for (int d = 0; d < 32; ++d) op[d] = (h16)(o[d] * rv);
  }
}

// ---------------------------------------------------------------------------
// Kernel 4: out = att @ proj_w + proj_b, fully async-fed (A and B tiles via
// GLOBAL_LOAD_ASYNC_TO_LDS_B128, 3 async instrs per wave per tile, 2-stage
// pipeline with s_wait_asynccnt 0x3). OOB A rows are clamped (results of
// those rows are never stored).
// ---------------------------------------------------------------------------
__global__ __launch_bounds__(256) void proj_gemm(
    const h16* __restrict__ att, const h16* __restrict__ wp_t,
    const float* __restrict__ bias, float* __restrict__ out)
{
  __shared__ __align__(16) h16 As[2][128][40];
  __shared__ __align__(16) h16 BsT[2][64][40];
  const int tid  = threadIdx.x;
  const int lane = tid & 31, wave = tid >> 5;
  const int r0 = blockIdx.y * 128;
  const int c0 = blockIdx.x * 64;
  const int m  = lane & 15, h = lane >> 4;

  // A issue: 128 rows x 32 halves = 8KB = 256 threads x two b128
  int arow0 = tid >> 1,           apart0 = tid & 1;
  int arow1 = (tid + 256) >> 1,   apart1 = (tid + 256) & 1;
  int rg0 = r0 + arow0; if (rg0 >= NROWS) rg0 = NROWS - 1;   // clamp: uniform issue
  int rg1 = r0 + arow1; if (rg1 >= NROWS) rg1 = NROWS - 1;
  const h16* asrc0 = att + (size_t)rg0 * C_ + apart0 * 16;
  const h16* asrc1 = att + (size_t)rg1 * C_ + apart1 * 16;
  const unsigned adst0[2] = {
    lds_off(&As[0][arow0][apart0 * 16]), lds_off(&As[1][arow0][apart0 * 16]) };
  const unsigned adst1[2] = {
    lds_off(&As[0][arow1][apart1 * 16]), lds_off(&As[1][arow1][apart1 * 16]) };
  // B issue: 64 rows x 32 halves = 4KB = 256 threads x one b128
  const int bc = tid >> 2, bseg = tid & 3;
  const unsigned bdst[2] = {
    lds_off(&BsT[0][bc][bseg * 8]), lds_off(&BsT[1][bc][bseg * 8]) };
  const h16* bsrc = wp_t + (size_t)(c0 + bc) * C_ + bseg * 8;

  v8f acc[4];
  for (int i = 0; i < 4; ++i) acc[i] = vzero8();

  // tile 0
  async_ld_b128(adst0[0], asrc0);
  async_ld_b128(adst1[0], asrc1);
  async_ld_b128(bdst[0], bsrc);

  for (int step = 0; step < 8; ++step) {
    const int kk0 = step * 32;
    __syncthreads();                       // prior buffer fully consumed
    if (step < 7) {
      const int nb = (step + 1) & 1;
      async_ld_b128(adst0[nb], asrc0 + kk0 + 32);
      async_ld_b128(adst1[nb], asrc1 + kk0 + 32);
      async_ld_b128(bdst[nb], bsrc + kk0 + 32);
      wait_async_le3();                    // tile `step` landed, t+1 in flight
    } else {
      wait_async_0();
    }
    __syncthreads();
    const int buf = step & 1;
    v16h a = cat8(ld8(&As[buf][wave * 16 + m][16 * h]),
                  ld8(&As[buf][wave * 16 + m][16 * h + 8]));
    for (int ci = 0; ci < 4; ++ci) {
      v16h bf = cat8(ld8(&BsT[buf][ci * 16 + m][16 * h]),
                     ld8(&BsT[buf][ci * 16 + m][16 * h + 8]));
      acc[ci] = wmma_f16(a, bf, acc[ci]);
    }
  }
  for (int r = 0; r < 8; ++r) {
    int row = r0 + wave * 16 + r + 8 * h;
    if (row >= NROWS) continue;
    int b = row / TPB, t = row - b * TPB;
    size_t obase = (t < NPATCH)
        ? (((size_t)b * NPATCH + t) * C_)
        : ((size_t)OUT1 + ((size_t)b * MDET + (t - NPATCH)) * C_);
    for (int ci = 0; ci < 4; ++ci) {
      int col = c0 + ci * 16 + m;
      out[obase + col] = acc[ci][r] + bias[col];
    }
  }
}

// ---------------------------------------------------------------------------
extern "C" void kernel_launch(void* const* d_in, const int* in_sizes, int n_in,
                              void* d_out, int out_size, void* d_ws, size_t ws_size,
                              hipStream_t stream) {
  const float* x      = (const float*)d_in[0];
  const float* det    = (const float*)d_in[1];
  const float* mask   = (const float*)d_in[2];
  const float* qkv_w  = (const float*)d_in[3];
  const float* qkv_b  = (const float*)d_in[4];
  const float* proj_w = (const float*)d_in[5];
  const float* proj_b = (const float*)d_in[6];
  const float* rpb    = (const float*)d_in[7];
  float* out = (float*)d_out;

  h16* qkv  = (h16*)d_ws;                       // NROWS * 768 f16 (~155 MB)
  h16* att  = qkv + (size_t)NROWS * TC;         // NROWS * 256 f16 (~52 MB)
  h16* wq_t = att + (size_t)NROWS * C_;         // 768*256 f16 transposed
  h16* wp_t = wq_t + (size_t)TC * C_;           // 256*256 f16 transposed

  prep_weights<<<(TC * C_ + C_ * C_ + 255) / 256, 256, 0, stream>>>(
      qkv_w, proj_w, wq_t, wp_t);

  dim3 g1(12, 791);
  qkv_gemm<<<g1, 256, 0, stream>>>(x, det, wq_t, qkv_b, qkv);

  win_attn<<<1568 * 4, 64, 0, stream>>>(qkv, mask, rpb, att);

  det_attn<<<64, 128, 0, stream>>>(qkv, att);

  dim3 g4(4, 791);
  proj_gemm<<<g4, 256, 0, stream>>>(att, wp_t, proj_b, out);
}
